// FrequencyWindowAttention_39496519254578
// MI455X (gfx1250) — matrix operand order repaired and checked by reading, not compiled, not run on hardware
//
#include <hip/hip_runtime.h>
#include <hip/hip_bf16.h>
#include <stdint.h>

typedef __attribute__((ext_vector_type(16))) _Float16 v16h;
typedef __attribute__((ext_vector_type(8)))  _Float16 v8h;
typedef __attribute__((ext_vector_type(4)))  _Float16 v4h;
typedef __attribute__((ext_vector_type(8)))  float    v8f;

#define WMMA_F16(a, b, c) \
  __builtin_amdgcn_wmma_f32_16x16x32_f16(false, (a), false, (b), (short)0, (c), false, false)

// Async global->LDS copy of 16 bytes per lane (GV mode, ASYNCcnt-tracked).
static __device__ inline void async_copy_b128(uint32_t lds_addr, uint64_t gaddr) {
  asm volatile("global_load_async_to_lds_b128 %0, %1, off"
               :: "v"(lds_addr), "v"(gaddr) : "memory");
}
static __device__ inline void wait_async0() {
  asm volatile("s_wait_asynccnt 0" ::: "memory");
}
static __device__ inline uint32_t lds_off(const void* p) {
  return (uint32_t)(uintptr_t)p;   // generic LDS pointer: addr[31:0] is the LDS byte address
}

// A-operand (16x32, f16, row-major source with stride in halves).
// Lane (m = lane&15) holds row m; 16 contiguous K values at (lane>>4)*16.
static __device__ inline v16h load_A16(const _Float16* __restrict__ base, int lane, int strideH) {
  int m  = lane & 15;
  int kh = (lane >> 4) << 4;                 // 0 or 16
  const _Float16* p = base + m * strideH + kh;
  v8h a0 = *(const v8h*)p;
  v8h a1 = *(const v8h*)(p + 8);
  v16h r;
#pragma unroll
  for (int i = 0; i < 8; ++i) { r[i] = a0[i]; r[i + 8] = a1[i]; }
  return r;
}

// B-operand (32x16, f16, column-major source = B^T row-major, stride in halves).
// Lane (n = lane&15) holds column n; 16 contiguous K values at (lane>>4)*16.
static __device__ inline v16h load_B16(const _Float16* __restrict__ bt, int lane, int strideH) {
  int n  = lane & 15;
  int kh = (lane >> 4) << 4;                 // 0 or 16
  return *(const v16h*)(bt + n * strideH + kh);
}

// ---------------------------------------------------------------- prep weights
// WqkvT[n*512+k] = (n<512 ? Wq[k][n] : Wkv[k][n-512]), WpT[n*512+k] = Wp[k][n]
__global__ __launch_bounds__(256) void k_prep_w(
    const float* __restrict__ Wq, const float* __restrict__ Wkv,
    const float* __restrict__ Wp,
    _Float16* __restrict__ WqkvT, _Float16* __restrict__ WpT) {
  int idx = blockIdx.x * 256 + threadIdx.x;
  if (idx < 1536 * 512) {
    int n = idx >> 9, k = idx & 511;
    float v = (n < 512) ? Wq[k * 512 + n] : Wkv[k * 1024 + (n - 512)];
    WqkvT[idx] = (_Float16)v;
  } else {
    int j = idx - 1536 * 512;                // < 512*512
    int n = j >> 9, k = j & 511;
    WpT[j] = (_Float16)Wp[k * 512 + n];
  }
}

// ---------------------------------------------------------------- bias + mask
// biasF[(h*192+q)*192+k] = tables[(lq*3+lk)*225 + rel(nq,nk)][h] + (lq!=lk ? -100 : 0)
__global__ __launch_bounds__(256) void k_bias(const float* __restrict__ bt,
                                              float* __restrict__ biasF) {
  int idx = blockIdx.x * 256 + threadIdx.x;  // < 8*192*192
  int h   = idx / (192 * 192);
  int rem = idx % (192 * 192);
  int q = rem / 192, k = rem % 192;
  int lq = q >> 6, nq = q & 63, lk = k >> 6, nk = k & 63;
  int iq = nq >> 3, jq = nq & 7, ik = nk >> 3, jk = nk & 7;
  int rel = (iq - ik + 7) * 15 + (jq - jk + 7);
  float v = bt[((lq * 3 + lk) * 225 + rel) * 8 + h];
  if (lq != lk) v += -100.0f;
  biasF[idx] = v;
}

// ---------------------------------------------------------------- QKV GEMM
// x(98304x512 f32) @ [Wq|Wkv](512x1536) -> Q,K (regrouped, row-major d), V^T ([d][key])
__global__ __launch_bounds__(256) void k_qkv(
    const float* __restrict__ x, const _Float16* __restrict__ WqkvT,
    const float* __restrict__ bq, const float* __restrict__ bkv,
    _Float16* __restrict__ Qd, _Float16* __restrict__ Kd, _Float16* __restrict__ VTd) {
  __shared__ __align__(32) _Float16 As[64 * 512];
  int tid  = threadIdx.x;
  int b    = blockIdx.y;                     // 0..1535 (one 64-row panel == one window-batch)
  int nt64 = blockIdx.x;                     // 0..23

  // stage A panel: 64x512 f32 -> f16 LDS (needs conversion, so no async path here)
  const float4* xs = (const float4*)(x + (size_t)b * 32768);
#pragma unroll
  for (int i = 0; i < 32; ++i) {
    int fi = tid + i * 256;                  // float4 index, 0..8191
    float4 v = xs[fi];
    v4h hv = { (_Float16)v.x, (_Float16)v.y, (_Float16)v.z, (_Float16)v.w };
    *(v4h*)(As + fi * 4) = hv;
  }
  __syncthreads();

  int w = tid >> 5, lane = tid & 31;
  int mt = w & 3;                            // 4 row tiles of 16
  int nh = w >> 2;                           // 2 col halves of 32
  int colbase = nt64 * 64 + nh * 32;
  v8f acc0 = {}; v8f acc1 = {};
#pragma unroll 4
  for (int ks = 0; ks < 16; ++ks) {
    v16h a  = load_A16(As + mt * 16 * 512 + ks * 32, lane, 512);
    v16h b0 = load_B16(WqkvT + (size_t)colbase * 512 + ks * 32, lane, 512);
    v16h b1 = load_B16(WqkvT + (size_t)(colbase + 16) * 512 + ks * 32, lane, 512);
    acc0 = WMMA_F16(a, b0, acc0);
    acc1 = WMMA_F16(a, b1, acc1);
  }

  int nl = lane & 15, hi = lane >> 4;
  int l = b >> 9, bnw = b & 511;
#pragma unroll
  for (int t = 0; t < 2; ++t) {
    const v8f& acc = t ? acc1 : acc0;
    int cg = colbase + t * 16 + nl;
    float bias = (cg < 512) ? bq[cg] : bkv[cg - 512];
    float sc   = (cg < 512) ? 0.125f : 1.0f;   // SCALE = 64^-0.5 applied to Q
#pragma unroll
    for (int r = 0; r < 8; ++r) {
      int m = mt * 16 + hi * 8 + r;            // n-position within window
      _Float16 hv = (_Float16)((acc[r] + bias) * sc);
      if (cg < 1024) {
        int h = (cg >> 6) & 7;
        int d = cg & 63;
        _Float16* dst = (cg < 512) ? Qd : Kd;
        dst[(size_t)((bnw * 8 + h) * 192 + l * 64 + m) * 64 + d] = hv;
      } else {
        int c = cg - 1024;
        int h = c >> 6, d = c & 63;
        VTd[(size_t)((bnw * 8 + h) * 64 + d) * 192 + (l * 64 + m)] = hv;
      }
    }
  }
}

// ---------------------------------------------------------------- attention
// one block = (bnw, head, 32-query row block): S=QK^T +bias, softmax, O=PV
__global__ __launch_bounds__(256) void k_attn(
    const _Float16* __restrict__ Qd, const _Float16* __restrict__ Kd,
    const _Float16* __restrict__ VTd, const float* __restrict__ biasF,
    _Float16* __restrict__ AOd) {
  __shared__ __align__(32) _Float16 Qs[32 * 64];
  __shared__ __align__(32) _Float16 Ks[192 * 64];
  __shared__ __align__(32) float    Ss[32 * 192];
  __shared__ __align__(32) _Float16 Ps[32 * 192];
  __shared__ float red[32 * 8];
  __shared__ float rowv[32];

  int tid = threadIdx.x;
  int rb  = blockIdx.x;                      // 0..5 (query row block)
  int bh  = blockIdx.y;                      // 0..4095
  int bnw = bh >> 3, h = bh & 7;
  size_t qkBase = (size_t)bh * 192 * 64;

  // async-stage 32x64 Q tile (4KB, 1 x b128/lane) and 192x64 K panel (24KB, 6 x b128/lane)
  {
    uint32_t qlds = lds_off(Qs) + (uint32_t)tid * 16;
    uint64_t qg   = (uint64_t)(uintptr_t)(Qd + qkBase + (size_t)rb * 32 * 64) + (uint64_t)tid * 16;
    async_copy_b128(qlds, qg);
    uint32_t klds = lds_off(Ks) + (uint32_t)tid * 16;
    uint64_t kg   = (uint64_t)(uintptr_t)(Kd + qkBase) + (uint64_t)tid * 16;
#pragma unroll
    for (int i = 0; i < 6; ++i)
      async_copy_b128(klds + (uint32_t)i * 4096, kg + (uint64_t)i * 4096);
    wait_async0();
  }
  __syncthreads();

  int w = tid >> 5, lane = tid & 31;
  int mt  = w & 1;                           // 2 row tiles of 16
  int cg4 = w >> 1;                          // 4 key groups of 48
  v8f acc[3] = {{}, {}, {}};
#pragma unroll
  for (int ks = 0; ks < 2; ++ks) {
    v16h a = load_A16(Qs + mt * 16 * 64 + ks * 32, lane, 64);
#pragma unroll
    for (int j = 0; j < 3; ++j) {
      int kb = cg4 * 48 + j * 16;
      v16h bm = load_B16(Ks + (size_t)kb * 64 + ks * 32, lane, 64);
      acc[j] = WMMA_F16(a, bm, acc[j]);
    }
  }
  int nl = lane & 15, hi = lane >> 4;
#pragma unroll
  for (int j = 0; j < 3; ++j) {
    int key = cg4 * 48 + j * 16 + nl;
#pragma unroll
    for (int r = 0; r < 8; ++r) {
      int m = mt * 16 + hi * 8 + r;
      int q = rb * 32 + m;
      Ss[m * 192 + key] = acc[j][r] + biasF[(size_t)(h * 192 + q) * 192 + key];
    }
  }
  __syncthreads();

  // softmax over 192 keys: 8 threads per row, 24 cols each
  int row = tid >> 3, seg = tid & 7;
  float* srow = Ss + row * 192 + seg * 24;
  float mx = -1e30f;
#pragma unroll
  for (int c = 0; c < 24; ++c) mx = fmaxf(mx, srow[c]);
  red[row * 8 + seg] = mx;
  __syncthreads();
  if (seg == 0) {
    float m2 = red[row * 8];
#pragma unroll
    for (int i = 1; i < 8; ++i) m2 = fmaxf(m2, red[row * 8 + i]);
    rowv[row] = m2;
  }
  __syncthreads();
  float rm = rowv[row];
  float s = 0.0f;
#pragma unroll
  for (int c = 0; c < 24; ++c) { float e = __expf(srow[c] - rm); srow[c] = e; s += e; }
  red[row * 8 + seg] = s;
  __syncthreads();
  if (seg == 0) {
    float t2 = 0.0f;
#pragma unroll
    for (int i = 0; i < 8; ++i) t2 += red[row * 8 + i];
    rowv[row] = t2;
  }
  __syncthreads();
  float inv = 1.0f / rowv[row];
  _Float16* prow = Ps + row * 192 + seg * 24;
#pragma unroll
  for (int c = 0; c < 24; ++c) prow[c] = (_Float16)(srow[c] * inv);
  __syncthreads();

  // O = P(32x192) @ V(192x64) using pre-transposed V
  int ntd = w >> 1;                          // 4 d-tiles of 16
  v8f oacc = {};
#pragma unroll
  for (int kk = 0; kk < 6; ++kk) {
    v16h a  = load_A16(Ps + mt * 16 * 192 + kk * 32, lane, 192);
    v16h bm = load_B16(VTd + (size_t)bh * 64 * 192 + (size_t)(ntd * 16) * 192 + kk * 32,
                       lane, 192);
    oacc = WMMA_F16(a, bm, oacc);
  }
#pragma unroll
  for (int r = 0; r < 8; ++r) {
    int m = mt * 16 + hi * 8 + r;
    int q = rb * 32 + m;
    int l = q >> 6, np = q & 63;
    int d = ntd * 16 + nl;
    size_t rg = (size_t)((l * 512 + bnw) * 64 + np);     // ungrouped row
    AOd[rg * 512 + h * 64 + d] = (_Float16)oacc[r];
  }
}

// ---------------------------------------------------------------- output proj
__global__ __launch_bounds__(256) void k_proj(
    const _Float16* __restrict__ AOd, const _Float16* __restrict__ WpT,
    const float* __restrict__ bp, float* __restrict__ out) {
  __shared__ __align__(32) _Float16 As[64 * 512];
  int tid = threadIdx.x;
  int p   = blockIdx.y;                      // 0..1535 row panel
  int nt  = blockIdx.x;                      // 0..7

  // async-stage the 64x512 f16 A panel (64KB, 16 x b128/lane)
  {
    uint32_t alds = lds_off(As) + (uint32_t)tid * 16;
    uint64_t ag   = (uint64_t)(uintptr_t)(AOd + (size_t)p * 64 * 512) + (uint64_t)tid * 16;
#pragma unroll
    for (int i = 0; i < 16; ++i)
      async_copy_b128(alds + (uint32_t)i * 4096, ag + (uint64_t)i * 4096);
    wait_async0();
  }
  __syncthreads();

  int w = tid >> 5, lane = tid & 31;
  int mt = w & 3, nh = w >> 2;
  int colbase = nt * 64 + nh * 32;
  v8f acc0 = {}, acc1 = {};
#pragma unroll 4
  for (int ks = 0; ks < 16; ++ks) {
    v16h a  = load_A16(As + mt * 16 * 512 + ks * 32, lane, 512);
    v16h b0 = load_B16(WpT + (size_t)colbase * 512 + ks * 32, lane, 512);
    v16h b1 = load_B16(WpT + (size_t)(colbase + 16) * 512 + ks * 32, lane, 512);
    acc0 = WMMA_F16(a, b0, acc0);
    acc1 = WMMA_F16(a, b1, acc1);
  }
  int nl = lane & 15, hi = lane >> 4;
#pragma unroll
  for (int t = 0; t < 2; ++t) {
    const v8f& acc = t ? acc1 : acc0;
    int cg = colbase + t * 16 + nl;
    float bias = bp[cg];
#pragma unroll
    for (int r = 0; r < 8; ++r) {
      int m = mt * 16 + hi * 8 + r;
      out[(size_t)(p * 64 + m) * 512 + cg] = acc[r] + bias;
    }
  }
}

// ---------------------------------------------------------------- launch
extern "C" void kernel_launch(void* const* d_in, const int* in_sizes, int n_in,
                              void* d_out, int out_size, void* d_ws, size_t ws_size,
                              hipStream_t stream) {
  const float* x    = (const float*)d_in[0];
  const float* Wq   = (const float*)d_in[1];
  const float* bq   = (const float*)d_in[2];
  const float* Wkv  = (const float*)d_in[3];
  const float* bkv  = (const float*)d_in[4];
  const float* btab = (const float*)d_in[5];
  const float* Wp   = (const float*)d_in[6];
  const float* bp   = (const float*)d_in[7];
  (void)in_sizes; (void)n_in; (void)out_size; (void)ws_size;

  char* ws = (char*)d_ws;
  size_t off = 0;
  auto alloc = [&](size_t bytes) { size_t o = off; off += (bytes + 255) & ~(size_t)255; return o; };
  _Float16* WqkvT = (_Float16*)(ws + alloc((size_t)1536 * 512 * 2));
  _Float16* WpT   = (_Float16*)(ws + alloc((size_t)512 * 512 * 2));
  float*    biasF = (float*)   (ws + alloc((size_t)8 * 192 * 192 * 4));
  _Float16* Qd    = (_Float16*)(ws + alloc((size_t)512 * 8 * 192 * 64 * 2));
  _Float16* Kd    = (_Float16*)(ws + alloc((size_t)512 * 8 * 192 * 64 * 2));
  _Float16* VTd   = (_Float16*)(ws + alloc((size_t)512 * 8 * 64 * 192 * 2));
  _Float16* AOd   = (_Float16*)(ws + alloc((size_t)98304 * 512 * 2));

  k_prep_w<<<4096, 256, 0, stream>>>(Wq, Wkv, Wp, WqkvT, WpT);
  k_bias<<<(8 * 192 * 192) / 256, 256, 0, stream>>>(btab, biasF);
  k_qkv<<<dim3(24, 1536), 256, 0, stream>>>(x, WqkvT, bq, bkv, Qd, Kd, VTd);
  k_attn<<<dim3(6, 4096), 256, 0, stream>>>(Qd, Kd, VTd, biasF, AOd);
  k_proj<<<dim3(8, 1536), 256, 0, stream>>>(AOd, WpT, bp, (float*)d_out);
}
